// TemporalGraphAttention_58909771432785
// MI455X (gfx1250) — matrix operand order repaired and checked by reading, not compiled, not run on hardware
//
#include <hip/hip_runtime.h>
#include <hip/hip_bf16.h>

typedef __attribute__((ext_vector_type(16))) _Float16 v16h;
typedef __attribute__((ext_vector_type(8)))  _Float16 h8;
typedef __attribute__((ext_vector_type(2)))  _Float16 h2;
typedef __attribute__((ext_vector_type(8)))  float    v8f;

#define NTOT   8192
#define NDIM   48
#define HID    64
#define HEADS  4
#define HD     16
#define ODIM   32

// ---- CDNA5 async global->LDS copy helpers (ASYNCcnt-tracked) --------------
__device__ __forceinline__ void async_ld_b128(void* lds, const void* g) {
    unsigned l = (unsigned)(unsigned long long)lds;          // LDS byte offset
    unsigned long long ga = (unsigned long long)g;
    asm volatile("global_load_async_to_lds_b128 %0, %1, off"
                 :: "v"(l), "v"(ga) : "memory");
}
__device__ __forceinline__ void async_ld_b32(void* lds, const void* g) {
    unsigned l = (unsigned)(unsigned long long)lds;
    unsigned long long ga = (unsigned long long)g;
    asm volatile("global_load_async_to_lds_b32 %0, %1, off"
                 :: "v"(l), "v"(ga) : "memory");
}

// ---------------------------------------------------------------------------
// Pass 1: enhanced[n][64] = concat(nodes[n][48], sin/cos time encoding)
// ---------------------------------------------------------------------------
__global__ __launch_bounds__(256) void k_enhance(const float* __restrict__ nodes,
                                                 const float* __restrict__ ts,
                                                 const float* __restrict__ freq,
                                                 float* __restrict__ enh) {
    int idx = blockIdx.x * 256 + threadIdx.x;        // NTOT*64 threads
    int n = idx >> 6, c = idx & 63;
    float v;
    if (c < NDIM) {
        v = nodes[n * NDIM + c];
    } else {
        int t = c - NDIM;
        float tt = ts[n];
        if (t < 8) v = __sinf(tt * freq[t]);
        else       v = __cosf(tt * freq[t - 8]);
    }
    enh[idx] = v;
}

// ---------------------------------------------------------------------------
// Pass 2: q,k (f16 [H][N][16]) and V transposed (f16 [H][16][N])
// ---------------------------------------------------------------------------
__global__ __launch_bounds__(256) void k_proj(const float* __restrict__ enh,
                                              const float* __restrict__ Wq,
                                              const float* __restrict__ Wk,
                                              const float* __restrict__ Wv,
                                              const float* __restrict__ bq,
                                              const float* __restrict__ bk,
                                              const float* __restrict__ bv,
                                              _Float16* __restrict__ qbuf,
                                              _Float16* __restrict__ kbuf,
                                              _Float16* __restrict__ vT) {
    int idx = blockIdx.x * 256 + threadIdx.x;        // HEADS*NTOT*HD threads
    int h   = idx >> 17;                             // / (8192*16)
    int rem = idx & (NTOT * HD - 1);
    int n = rem >> 4, d = rem & 15;
    const float* er = enh + (size_t)n * HID;
    const float* wq = Wq + (size_t)h * HID * HD + d;
    const float* wk = Wk + (size_t)h * HID * HD + d;
    const float* wv = Wv + (size_t)h * HID * HD + d;
    float aq = 0.f, ak = 0.f, av = 0.f;
#pragma unroll 8
    for (int e = 0; e < HID; ++e) {
        float x = er[e];
        aq = fmaf(x, wq[e * HD], aq);
        ak = fmaf(x, wk[e * HD], ak);
        av = fmaf(x, wv[e * HD], av);
    }
    aq += bq[h * HD + d];
    ak += bk[h * HD + d];
    av += bv[h * HD + d];
    qbuf[((size_t)h * NTOT + n) * HD + d] = (_Float16)aq;
    kbuf[((size_t)h * NTOT + n) * HD + d] = (_Float16)ak;
    vT[((size_t)h * HD + d) * NTOT + n]   = (_Float16)av;
}

// ---------------------------------------------------------------------------
// Pass 3: fused flash attention with decay + adjacency mask.
// Block = 256 threads = 8 waves; wave w -> head = w&3, row-half rb = w>>2.
// 32 query rows per block, chunks of 32 keys, double-buffered async staging.
// ---------------------------------------------------------------------------
__global__ __launch_bounds__(256) void k_attn(const _Float16* __restrict__ qbuf,
                                              const _Float16* __restrict__ kbuf,
                                              const _Float16* __restrict__ vT,
                                              const float* __restrict__ adj,
                                              const float* __restrict__ ts,
                                              float* __restrict__ Obuf) {
    __shared__ alignas(16) float     adjT[2][32][32];            // 8 KB (dbuf)
    __shared__ alignas(16) float     dlds[32][32];               // 4 KB decay
    __shared__ float                 tsi[32];
    __shared__ float                 tsj[2][32];
    __shared__ alignas(64) _Float16  klds[2][HEADS][32][HD];     // 8 KB (dbuf)
    __shared__ alignas(64) _Float16  vtlds[2][HEADS][HD][32];    // 8 KB (dbuf)
    __shared__ alignas(16) float     slds[2][HEADS][16][32];     // 16 KB (rb)
    __shared__ alignas(16) _Float16  plds[2][HEADS][16][32];     // 8 KB (rb)
    __shared__ float                 corrl[2][HEADS][16];
    __shared__ float                 lsum[2][HEADS][16];

    const int tid  = threadIdx.x;
    const int lane = tid & 31;
    const int wave = tid >> 5;
    const int head = wave & 3;
    const int rb   = wave >> 2;
    const int hi   = lane >> 4;
    const int ln16 = lane & 15;
    const int i0   = blockIdx.x * 32;

    if (tid < 32) tsi[tid] = ts[i0 + tid];

    // issue async staging of one 32-key chunk into buffer b
    // (exactly 4 async instructions per wave -> uniform ASYNCcnt)
    auto stage = [&](int b, int j0) {
        int r = tid >> 3, c4 = (tid & 7) << 2;
        async_ld_b128(&adjT[b][r][c4], adj + (size_t)(i0 + r) * NTOT + j0 + c4);
        int base = tid * 8;
        int h = base >> 9, rem = base & 511;
        int jk = rem >> 4, d0 = rem & 15;
        async_ld_b128(&klds[b][h][jk][d0],
                      kbuf + ((size_t)h * NTOT + j0 + jk) * HD + d0);
        int dv = rem >> 5, jo = rem & 31;
        async_ld_b128(&vtlds[b][h][dv][jo],
                      vT + ((size_t)h * HD + dv) * NTOT + j0 + jo);
        if (lane < 4)
            async_ld_b32(&tsj[b][wave * 4 + lane], ts + j0 + wave * 4 + lane);
    };

    // Q fragment (A layout, 16x32, real K = head dims 0..15, K 16..31 zero pad)
    v16h qa;
#pragma unroll
    for (int t = 0; t < 16; ++t) qa[t] = (_Float16)0;
    {
        int row = i0 + rb * 16 + ln16;
        if (hi == 0) qa = *(const v16h*)(qbuf + ((size_t)head * NTOT + row) * HD);
    }

    float m_run = -1e30f, l_run = 0.f;
    v8f oacc;
#pragma unroll
    for (int t = 0; t < 8; ++t) oacc[t] = 0.f;

    stage(0, 0);                                 // prefetch first chunk

    const int NCH = NTOT / 32;
    for (int jc = 0; jc < NCH; ++jc) {
        const int cur = jc & 1;
        const int j0  = jc * 32;
        __syncthreads();            // prev iteration consumers of buf[1-cur] done

        if (jc + 1 < NCH) {         // pipeline next chunk, then drain current
            stage(1 - cur, j0 + 32);
            asm volatile("s_wait_asynccnt 0x4" ::: "memory");
        } else {
            asm volatile("s_wait_asynccnt 0x0" ::: "memory");
        }
        __syncthreads();            // all waves' staging for chunk jc visible

        // ---- shared decay tile: one exp per (i,j), reused by all 4 heads ----
#pragma unroll
        for (int k = 0; k < 4; ++k) {
            int idx = tid * 4 + k;
            int r = idx >> 5, c = idx & 31;
            dlds[r][c] = __expf(-fabsf(tsi[r] - tsj[cur][c]) * 0.1f);
        }
        __syncthreads();            // dlds visible to all waves

        // ---- scores: two 16x16 WMMAs (K padded 16->32) ----
        v8f s0, s1;
#pragma unroll
        for (int t = 0; t < 8; ++t) { s0[t] = 0.f; s1[t] = 0.f; }
        v16h kb0, kb1;
        if (hi == 0) {
            kb0 = *(const v16h*)&klds[cur][head][ln16][0];
            kb1 = *(const v16h*)&klds[cur][head][16 + ln16][0];
        } else {
#pragma unroll
            for (int t = 0; t < 16; ++t) { kb0[t] = (_Float16)0; kb1[t] = (_Float16)0; }
        }
        s0 = __builtin_amdgcn_wmma_f32_16x16x32_f16(false, qa, false, kb0,
                                                    (short)0, s0, false, false);
        s1 = __builtin_amdgcn_wmma_f32_16x16x32_f16(false, qa, false, kb1,
                                                    (short)0, s1, false, false);
#pragma unroll
        for (int v = 0; v < 8; ++v) {
            int m = v + 8 * hi;
            slds[rb][head][m][ln16]      = s0[v];
            slds[rb][head][m][16 + ln16] = s1[v];
        }
        // slds/plds/corrl are wave-private: DS-count fence instead of barrier
        asm volatile("s_wait_dscnt 0x0" ::: "memory");

        // ---- flash softmax: lane pair (r, r+16) splits row r's 32 columns ----
        {
            const int r  = ln16;            // row within this wave's 16-row tile
            const int hc = hi;              // which 16-column half this lane owns
            const int gi = i0 + rb * 16 + r;
            float sv[16];
            float mx = m_run;
#pragma unroll
            for (int c4 = 0; c4 < 4; ++c4) {
                float4 s4 = *(const float4*)&slds[rb][head][r][hc * 16 + c4 * 4];
                float ss[4] = {s4.x, s4.y, s4.z, s4.w};
#pragma unroll
                for (int k = 0; k < 4; ++k) {
                    int c = hc * 16 + c4 * 4 + k;
                    float val = ss[k] * 0.25f * dlds[rb * 16 + r][c];
                    bool ok = (adjT[cur][rb * 16 + r][c] > 0.f) || (gi == (j0 + c));
                    val = ok ? val : -1e30f;
                    sv[c4 * 4 + k] = val;
                    mx = fmaxf(mx, val);
                }
            }
            mx = fmaxf(mx, __shfl_xor(mx, 16, 32));   // combine halves
            float corr = __expf(m_run - mx);          // sentinel keeps finite
            m_run = mx;
            float addl = 0.f;
#pragma unroll
            for (int c = 0; c < 16; c += 2) {
                float p0 = (sv[c]     > -1e29f) ? __expf(sv[c]     - mx) : 0.f;
                float p1 = (sv[c + 1] > -1e29f) ? __expf(sv[c + 1] - mx) : 0.f;
                addl += p0 + p1;
                h2 pp;
                pp[0] = (_Float16)p0;
                pp[1] = (_Float16)p1;
                *(h2*)&plds[rb][head][r][hc * 16 + c] = pp;
            }
            addl += __shfl_xor(addl, 16, 32);         // both lanes: full row sum
            l_run = l_run * corr + addl;
            if (hc == 0) corrl[rb][head][r] = corr;
        }
        asm volatile("s_wait_dscnt 0x0" ::: "memory");

        // ---- rescale accumulator, then P(16x32) x V(32x16) WMMA ----
#pragma unroll
        for (int v = 0; v < 8; ++v) oacc[v] *= corrl[rb][head][v + 8 * hi];

        v16h pa, vb;
        {
            h8 plo = *(const h8*)&plds[rb][head][ln16][16 * hi];
            h8 phi = *(const h8*)&plds[rb][head][ln16][16 * hi + 8];
#pragma unroll
            for (int t = 0; t < 8; ++t) { pa[t] = plo[t]; pa[8 + t] = phi[t]; }
            vb = *(const v16h*)&vtlds[cur][head][ln16][16 * hi];
        }
        oacc = __builtin_amdgcn_wmma_f32_16x16x32_f16(false, pa, false, vb,
                                                      (short)0, oacc, false, false);
    }

    // ---- finalize: divide by l and write mh (= concat of heads) ----
    if (hi == 0) lsum[rb][head][ln16] = l_run;
    asm volatile("s_wait_dscnt 0x0" ::: "memory");
#pragma unroll
    for (int v = 0; v < 8; ++v) {
        int m = v + 8 * hi;
        int row = i0 + rb * 16 + m;
        Obuf[(size_t)row * HID + head * HD + ln16] = oacc[v] / lsum[rb][head][m];
    }
}

// ---------------------------------------------------------------------------
// Pass 4: residual + LayerNorm + GELU FFN. One wave per node row.
// ---------------------------------------------------------------------------
__global__ __launch_bounds__(256) void k_ffn(const float* __restrict__ Obuf,
                                             const float* __restrict__ enh,
                                             const float* __restrict__ lnS,
                                             const float* __restrict__ lnB,
                                             const float* __restrict__ W1,
                                             const float* __restrict__ b1,
                                             const float* __restrict__ W2,
                                             const float* __restrict__ b2,
                                             float* __restrict__ out) {
    __shared__ float ylds[8][HID];
    __shared__ float hlds[8][HID * 2];
    const int tid = threadIdx.x;
    const int w = tid >> 5, l = tid & 31;
    const int n = blockIdx.x * 8 + w;
    const int c0 = 2 * l;

    float x0 = Obuf[(size_t)n * HID + c0]     + enh[(size_t)n * HID + c0];
    float x1 = Obuf[(size_t)n * HID + c0 + 1] + enh[(size_t)n * HID + c0 + 1];
    float s = x0 + x1;
#pragma unroll
    for (int off = 16; off > 0; off >>= 1) s += __shfl_xor(s, off, 32);
    float mu = s * (1.0f / HID);
    float d0 = x0 - mu, d1 = x1 - mu;
    float vv = d0 * d0 + d1 * d1;
#pragma unroll
    for (int off = 16; off > 0; off >>= 1) vv += __shfl_xor(vv, off, 32);
    float inv = rsqrtf(vv * (1.0f / HID) + 1e-6f);
    ylds[w][c0]     = d0 * inv * lnS[c0]     + lnB[c0];
    ylds[w][c0 + 1] = d1 * inv * lnS[c0 + 1] + lnB[c0 + 1];
    __syncthreads();

    float acc[4];
#pragma unroll
    for (int k = 0; k < 4; ++k) acc[k] = b1[l * 4 + k];
#pragma unroll 8
    for (int e = 0; e < HID; ++e) {
        float ye = ylds[w][e];
#pragma unroll
        for (int k = 0; k < 4; ++k)
            acc[k] = fmaf(ye, W1[e * (HID * 2) + l * 4 + k], acc[k]);
    }
#pragma unroll
    for (int k = 0; k < 4; ++k) {
        float x = acc[k];
        float g = 0.5f * x * (1.0f + tanhf(0.7978845608f * (x + 0.044715f * x * x * x)));
        hlds[w][l * 4 + k] = g;
    }
    __syncthreads();

    float a = b2[l];
#pragma unroll 8
    for (int k2 = 0; k2 < HID * 2; ++k2)
        a = fmaf(hlds[w][k2], W2[k2 * ODIM + l], a);
    out[(size_t)n * ODIM + l] = a;
}

// ---------------------------------------------------------------------------
extern "C" void kernel_launch(void* const* d_in, const int* in_sizes, int n_in,
                              void* d_out, int out_size, void* d_ws, size_t ws_size,
                              hipStream_t stream) {
    (void)in_sizes; (void)n_in; (void)out_size; (void)ws_size;
    const float* nodes = (const float*)d_in[0];
    const float* adj   = (const float*)d_in[1];
    const float* ts    = (const float*)d_in[2];
    const float* freq  = (const float*)d_in[3];
    const float* Wq    = (const float*)d_in[4];
    const float* bq    = (const float*)d_in[5];
    const float* Wk    = (const float*)d_in[6];
    const float* bk    = (const float*)d_in[7];
    const float* Wv    = (const float*)d_in[8];
    const float* bv    = (const float*)d_in[9];
    const float* lnS   = (const float*)d_in[10];
    const float* lnB   = (const float*)d_in[11];
    const float* W1    = (const float*)d_in[12];
    const float* b1    = (const float*)d_in[13];
    const float* W2    = (const float*)d_in[14];
    const float* b2    = (const float*)d_in[15];

    char* ws = (char*)d_ws;                     // 7 MB used
    float*    enh  = (float*)ws;                         // [N][64] f32, 2 MB
    _Float16* qbuf = (_Float16*)(ws + (2u << 20));       // [H][N][16] f16, 1 MB
    _Float16* kbuf = (_Float16*)(ws + (3u << 20));       // [H][N][16] f16, 1 MB
    _Float16* vT   = (_Float16*)(ws + (4u << 20));       // [H][16][N] f16, 1 MB
    float*    Obuf = (float*)(ws + (5u << 20));          // [N][64] f32, 2 MB

    k_enhance<<<NTOT * HID / 256, 256, 0, stream>>>(nodes, ts, freq, enh);
    k_proj<<<HEADS * NTOT * HD / 256, 256, 0, stream>>>(enh, Wq, Wk, Wv, bq, bk, bv,
                                                        qbuf, kbuf, vT);
    k_attn<<<NTOT / 32, 256, 0, stream>>>(qbuf, kbuf, vT, adj, ts, Obuf);
    k_ffn<<<NTOT / 8, 256, 0, stream>>>(Obuf, enh, lnS, lnB, W1, b1, W2, b2,
                                        (float*)d_out);
}
